// NNLoss_23630910063047
// MI455X (gfx1250) — matrix operand order repaired and checked by reading, not compiled, not run on hardware
//
#include <hip/hip_runtime.h>
#include <hip/hip_bf16.h>

typedef _Float16 v8h  __attribute__((ext_vector_type(8)));
typedef _Float16 v16h __attribute__((ext_vector_type(16)));
typedef float    v8f  __attribute__((ext_vector_type(8)));
typedef int      v4i  __attribute__((ext_vector_type(4)));

#define CDIM 768
#define HW   16384
#define EPSF 1e-8f

// A-panel LDS row stride in halves (768 + 8 pad: 16B aligned, 388 dwords/row,
// 388 % 64 == 4 -> spreads banks)
#define KPAD  776
// B K-half chunk: 384 halves + 8 pad (196 dwords/row, 196 % 64 == 4)
#define KH    384
#define KHPAD 392

// ---------------------------------------------------------------------------
// CDNA5 async global->LDS copy (ASYNCcnt path), with sync fallback.
// Builtin params are pointers to int __vector(4) in AS1 (global) / AS3 (LDS).
// ---------------------------------------------------------------------------
#if defined(__has_builtin)
#if __has_builtin(__builtin_amdgcn_global_load_async_to_lds_b128)
#define HAVE_ASYNC_LDS 1
#endif
#endif

#ifdef HAVE_ASYNC_LDS
typedef __attribute__((address_space(1))) v4i g_v4i;
typedef __attribute__((address_space(3))) v4i l_v4i;
static __device__ __forceinline__ void async_cp16(const void* g, void* l) {
    __builtin_amdgcn_global_load_async_to_lds_b128((g_v4i*)g, (l_v4i*)l, 0, 0);
}
static __device__ __forceinline__ void async_wait0() {
    asm volatile("s_wait_asynccnt 0" ::: "memory");
}
#else
static __device__ __forceinline__ void async_wait0() {}
#endif

// Issue copy of `rows` rows x 768 halves (row stride 768) into LDS (stride KPAD).
static __device__ __forceinline__ void panel_issue_A(const _Float16* __restrict__ gsrc,
                                                     size_t grow0,
                                                     _Float16* __restrict__ ldst,
                                                     int rows, int t) {
#ifdef HAVE_ASYNC_LDS
    const int units = rows * (CDIM / 8);   // 16B units, 96 per row
    for (int idx = t; idx < units; idx += 256) {
        int r = idx / (CDIM / 8);
        int d = idx - r * (CDIM / 8);
        async_cp16(gsrc + (grow0 + r) * CDIM + d * 8,
                   ldst + r * KPAD + d * 8);
    }
#else
    const unsigned int* src = (const unsigned int*)gsrc;
    unsigned int* dst = (unsigned int*)ldst;
    for (int idx = t; idx < rows * (CDIM / 2); idx += 256) {
        int r = idx / (CDIM / 2);
        int d = idx - r * (CDIM / 2);
        dst[r * (KPAD / 2) + d] = src[(grow0 + r) * (CDIM / 2) + d];
    }
#endif
}

// Issue copy of one B chunk: 64 rows x 384 halves (K-half kh) into LDS buffer
// with row stride KHPAD. No wait here (pipelined).
static __device__ __forceinline__ void bchunk_issue(const _Float16* __restrict__ Bt,
                                                    int j0, int kh,
                                                    _Float16* __restrict__ buf, int t) {
#ifdef HAVE_ASYNC_LDS
    const int units = 64 * (KH / 8);   // 48 units per row, 3072 total
    for (int idx = t; idx < units; idx += 256) {
        int r = idx / (KH / 8);
        int d = idx - r * (KH / 8);
        async_cp16(Bt + (size_t)(j0 + r) * CDIM + kh * KH + d * 8,
                   buf + r * KHPAD + d * 8);
    }
#else
    const unsigned int* src = (const unsigned int*)Bt;
    unsigned int* dst = (unsigned int*)buf;
    for (int idx = t; idx < 64 * (KH / 2); idx += 256) {
        int r = idx / (KH / 2);
        int d = idx - r * (KH / 2);
        dst[r * (KHPAD / 2) + d] =
            src[(size_t)(j0 + r) * (CDIM / 2) + kh * (KH / 2) + d];
    }
#endif
}

// ---------------------------------------------------------------------------
// Kernel 1: per-channel means of a and b (reduce over 16384 positions)
// ---------------------------------------------------------------------------
__global__ void nn_mean_kernel(const float* __restrict__ a,
                               const float* __restrict__ b,
                               float* __restrict__ mean_a,
                               float* __restrict__ mean_b) {
    const int c = blockIdx.x;
    const int t = threadIdx.x;
    float sa = 0.f, sb = 0.f;
    const float* ar = a + (size_t)c * HW;
    const float* br = b + (size_t)c * HW;
    for (int i = t; i < HW; i += 256) { sa += ar[i]; sb += br[i]; }
    __shared__ float ra[256], rb[256];
    ra[t] = sa; rb[t] = sb;
    __syncthreads();
    for (int s = 128; s > 0; s >>= 1) {
        if (t < s) { ra[t] += ra[t + s]; rb[t] += rb[t + s]; }
        __syncthreads();
    }
    if (t == 0) {
        mean_a[c] = ra[0] * (1.0f / HW);
        mean_b[c] = rb[0] * (1.0f / HW);
    }
}

// ---------------------------------------------------------------------------
// Kernel 2: center + normalize (sqrt(ss+EPS)+EPS), pack to f16, TRANSPOSED
// output layout: out[i][c]; LDS-staged so global f16 stores are coalesced.
// ---------------------------------------------------------------------------
__global__ void nn_normpack_kernel(const float* __restrict__ x,
                                   const float* __restrict__ mean,
                                   _Float16* __restrict__ out) {
    const int t  = threadIdx.x;
    const int i0 = blockIdx.x * 256;
    const int i  = i0 + t;

    __shared__ float sm[CDIM];
    for (int c = t; c < CDIM; c += 256) sm[c] = mean[c];
    __syncthreads();

    float ss = 0.f;
    for (int c = 0; c < CDIM; ++c) {
        float v = x[(size_t)c * HW + i] - sm[c];
        ss += v * v;
    }
    const float scale = 1.0f / (sqrtf(ss + EPSF) + EPSF);

    __shared__ _Float16 tile[256 * 34];   // 32 halves/row + 2 pad (17 dwords)
    unsigned int* outu = (unsigned int*)out;
    for (int c0 = 0; c0 < CDIM; c0 += 32) {
        #pragma unroll
        for (int cc = 0; cc < 32; ++cc) {
            float v = (x[(size_t)(c0 + cc) * HW + i] - sm[c0 + cc]) * scale;
            tile[t * 34 + cc] = (_Float16)v;
        }
        __syncthreads();
        #pragma unroll
        for (int k2 = 0; k2 < 16; ++k2) {
            int flat = k2 * 256 + t;
            int r = flat >> 4, col = flat & 15;
            unsigned int d = *(const unsigned int*)&tile[r * 34 + col * 2];
            outu[(size_t)(i0 + r) * (CDIM / 2) + (c0 >> 1) + col] = d;
        }
        __syncthreads();
    }
}

// ---------------------------------------------------------------------------
// Kernel 3: fused WMMA GEMM (a^T b, K=768) + row argmax.
// 8 waves as 4(M) x 2(N); each wave owns a 32x32 register tile (4 accums)
// -> 2 ds_load_b128 per WMMA. A panel (128x768, 199KB) LDS-resident.
// B streamed as double-buffered (j-tile, K-half) chunks of 64x384 (50KB each):
// async loads for chunk i+1 overlap WMMA compute on chunk i (ASYNCcnt).
// ---------------------------------------------------------------------------
__global__ void __launch_bounds__(256)
nn_argmax_kernel(const _Float16* __restrict__ At,   // [HW][CDIM]
                 const _Float16* __restrict__ Bt,   // [HW][CDIM]
                 int* __restrict__ zbest) {
    extern __shared__ _Float16 lds[];
    _Float16* As  = lds;                          // [128][KPAD]
    _Float16* Bs0 = lds + 128 * KPAD;             // [64][KHPAD]
    _Float16* Bs1 = Bs0 + 64 * KHPAD;             // [64][KHPAD]

    const int t  = threadIdx.x;
    const int i0 = blockIdx.x * 128;

    panel_issue_A(At, (size_t)i0, As, 128, t);
    bchunk_issue(Bt, 0, 0, Bs0, t);
    async_wait0();
    __syncthreads();

    const int lane = t & 31;
    const int wv   = t >> 5;
    const int wm   = wv & 3;        // M sub-panel (32 rows each)
    const int wn   = wv >> 2;       // N half (32 cols each)
    const int half = lane >> 4;     // 0: lanes 0-15, 1: lanes 16-31
    const int ln   = lane & 15;

    float best0[8], best1[8];
    int   barg0[8], barg1[8];
    #pragma unroll
    for (int v = 0; v < 8; ++v) {
        best0[v] = -3.4e38f; barg0[v] = 0;
        best1[v] = -3.4e38f; barg1[v] = 0;
    }

    // A fragment bases (ISA 7.12.2): lane>=16 reads K+8 / K+24.
    const _Float16* arow0 = As + (32 * wm + ln) * KPAD + (half ? 8 : 0);
    const _Float16* arow1 = arow0 + 16 * KPAD;
    const int boff = (32 * wn + ln) * KHPAD + (half ? 8 : 0);

    v8f acc00 = {}, acc01 = {}, acc10 = {}, acc11 = {};

    const int numChunks = (HW / 64) * 2;   // (j-tile, K-half) pairs, ascending j
    for (int ch = 0; ch < numChunks; ++ch) {
        _Float16* cur = (ch & 1) ? Bs1 : Bs0;
        _Float16* nxt = (ch & 1) ? Bs0 : Bs1;
        if (ch + 1 < numChunks) {
            int nch = ch + 1;
            bchunk_issue(Bt, (nch >> 1) * 64, nch & 1, nxt, t);
        }

        const int kh = ch & 1;
        const _Float16* a0 = arow0 + kh * KH;
        const _Float16* a1 = arow1 + kh * KH;
        const _Float16* brow0 = cur + boff;
        const _Float16* brow1 = brow0 + 16 * KHPAD;

        #pragma unroll 2
        for (int k = 0; k < KH / 32; ++k) {
            const int ko = 32 * k;
            v8h a0lo = *(const v8h*)(a0 + ko);
            v8h a0hi = *(const v8h*)(a0 + ko + 16);
            v8h a1lo = *(const v8h*)(a1 + ko);
            v8h a1hi = *(const v8h*)(a1 + ko + 16);
            v8h b0lo = *(const v8h*)(brow0 + ko);
            v8h b0hi = *(const v8h*)(brow0 + ko + 16);
            v8h b1lo = *(const v8h*)(brow1 + ko);
            v8h b1hi = *(const v8h*)(brow1 + ko + 16);
            v16h af0 = __builtin_shufflevector(a0lo, a0hi,
                0,1,2,3,4,5,6,7,8,9,10,11,12,13,14,15);
            v16h af1 = __builtin_shufflevector(a1lo, a1hi,
                0,1,2,3,4,5,6,7,8,9,10,11,12,13,14,15);
            v16h bf0 = __builtin_shufflevector(b0lo, b0hi,
                0,1,2,3,4,5,6,7,8,9,10,11,12,13,14,15);
            v16h bf1 = __builtin_shufflevector(b1lo, b1hi,
                0,1,2,3,4,5,6,7,8,9,10,11,12,13,14,15);
            acc00 = __builtin_amdgcn_wmma_f32_16x16x32_f16(
                false, af0, false, bf0, (short)0, acc00, false, false);
            acc01 = __builtin_amdgcn_wmma_f32_16x16x32_f16(
                false, af0, false, bf1, (short)0, acc01, false, false);
            acc10 = __builtin_amdgcn_wmma_f32_16x16x32_f16(
                false, af1, false, bf0, (short)0, acc10, false, false);
            acc11 = __builtin_amdgcn_wmma_f32_16x16x32_f16(
                false, af1, false, bf1, (short)0, acc11, false, false);
        }

        if (kh == 1) {
            // Full K=768 accumulated for this j-tile: fold into running argmax.
            // D layout: lane n(<16): acc[v]=D[M=v][N=n]; lane n+16: D[M=8+v][N=n]
            // accX0 column is j0+32*wn+ln, accX1 is +16; ascending-j order with
            // strict '>' preserves first-argmax semantics.
            const int jj = (ch >> 1) * 64 + 32 * wn + ln;
            #pragma unroll
            for (int v = 0; v < 8; ++v) {
                if (acc00[v] > best0[v]) { best0[v] = acc00[v]; barg0[v] = jj; }
                if (acc01[v] > best0[v]) { best0[v] = acc01[v]; barg0[v] = jj + 16; }
                if (acc10[v] > best1[v]) { best1[v] = acc10[v]; barg1[v] = jj; }
                if (acc11[v] > best1[v]) { best1[v] = acc11[v]; barg1[v] = jj + 16; }
            }
            v8f z = {};
            acc00 = z; acc01 = z; acc10 = z; acc11 = z;
        }

        async_wait0();       // next chunk landed
        __syncthreads();     // everyone done reading cur before it is reused
    }

    // reduce over the 16 lanes holding each row; tie-break to smaller j
    #pragma unroll
    for (int v = 0; v < 8; ++v) {
        for (int off = 1; off < 16; off <<= 1) {
            float ov0 = __shfl_xor(best0[v], off, 32);
            int   oi0 = __shfl_xor(barg0[v], off, 32);
            if (ov0 > best0[v] || (ov0 == best0[v] && oi0 < barg0[v])) {
                best0[v] = ov0; barg0[v] = oi0;
            }
            float ov1 = __shfl_xor(best1[v], off, 32);
            int   oi1 = __shfl_xor(barg1[v], off, 32);
            if (ov1 > best1[v] || (ov1 == best1[v] && oi1 < barg1[v])) {
                best1[v] = ov1; barg1[v] = oi1;
            }
        }
    }

    // combine the two N-halves (waves wn=0/1 share rows) through LDS
    float* sval = (float*)Bs0;           // [2][128]
    int*   sidx = (int*)(Bs0 + 1024);    // [2][128] (2KB past sval)
    if (ln == 0) {
        #pragma unroll
        for (int v = 0; v < 8; ++v) {
            int r0 = 32 * wm + 8 * half + v;      // Mi = 0
            sval[wn * 128 + r0] = best0[v];
            sidx[wn * 128 + r0] = barg0[v];
            int r1 = r0 + 16;                     // Mi = 1
            sval[wn * 128 + r1] = best1[v];
            sidx[wn * 128 + r1] = barg1[v];
        }
    }
    __syncthreads();
    if (t < 128) {
        float v0 = sval[t], v1 = sval[128 + t];
        int   x0 = sidx[t], x1 = sidx[128 + t];
        int win = (v1 > v0 || (v1 == v0 && x1 < x0)) ? x1 : x0;
        zbest[i0 + t] = win;
    }
}

// ---------------------------------------------------------------------------
// Kernel 4: per-position cosine loss on ORIGINAL fp32 data, block partials
// ---------------------------------------------------------------------------
__global__ void nn_loss_partial_kernel(const float* __restrict__ a,
                                       const float* __restrict__ b,
                                       const int* __restrict__ z,
                                       float* __restrict__ partial) {
    const int t = threadIdx.x;
    const int i = blockIdx.x * 256 + t;
    const int zi = z[i];
    float na = 0.f, nb = 0.f, dot = 0.f;
    for (int c = 0; c < CDIM; ++c) {
        float av = a[(size_t)c * HW + i];
        float bv = b[(size_t)c * HW + zi];
        na += av * av; nb += bv * bv; dot += av * bv;
    }
    float cs  = dot / ((sqrtf(na) + EPSF) * (sqrtf(nb) + EPSF));
    __shared__ float red[256];
    red[t] = 1.0f - cs;
    __syncthreads();
    for (int s = 128; s > 0; s >>= 1) {
        if (t < s) red[t] += red[t + s];
        __syncthreads();
    }
    if (t == 0) partial[blockIdx.x] = red[0];
}

__global__ void nn_final_kernel(const float* __restrict__ partial,
                                float* __restrict__ out) {
    __shared__ float red[64];
    const int t = threadIdx.x;
    red[t] = partial[t];
    __syncthreads();
    for (int s = 32; s > 0; s >>= 1) {
        if (t < s) red[t] += red[t + s];
        __syncthreads();
    }
    if (t == 0) out[0] = red[0] * (1.0f / HW);
}

// ---------------------------------------------------------------------------
extern "C" void kernel_launch(void* const* d_in, const int* in_sizes, int n_in,
                              void* d_out, int out_size, void* d_ws, size_t ws_size,
                              hipStream_t stream) {
    const float* a = (const float*)d_in[0];   // x_feats (1,768,128,128) fp32
    const float* b = (const float*)d_in[1];   // s_feats

    char* ws = (char*)d_ws;
    const size_t packBytes = (size_t)HW * CDIM * sizeof(_Float16); // 24 MB
    _Float16* At     = (_Float16*)ws;
    _Float16* Bt     = (_Float16*)(ws + packBytes);
    float*    mean_a = (float*)(ws + 2 * packBytes);
    float*    mean_b = mean_a + CDIM;
    int*      zb     = (int*)(mean_b + CDIM);
    float*    part   = (float*)(zb + HW);

    nn_mean_kernel<<<CDIM, 256, 0, stream>>>(a, b, mean_a, mean_b);
    nn_normpack_kernel<<<HW / 256, 256, 0, stream>>>(a, mean_a, At);
    nn_normpack_kernel<<<HW / 256, 256, 0, stream>>>(b, mean_b, Bt);

    const size_t ldsBytes =
        (size_t)(128 * KPAD + 2 * 64 * KHPAD) * sizeof(_Float16);  // 299,008 B
    nn_argmax_kernel<<<HW / 128, 256, ldsBytes, stream>>>(At, Bt, zb);

    nn_loss_partial_kernel<<<HW / 256, 256, 0, stream>>>(a, b, zb, part);
    nn_final_kernel<<<1, 64, 0, stream>>>(part, (float*)d_out);
}
